// ffEIRNN_37606733643976
// MI455X (gfx1250) — compile-verified, matrix operand order checked by reading, not run on hardware
//
#include <hip/hip_runtime.h>
#include <hip/hip_bf16.h>

// Problem dims (fixed by the reference)
#define T_DIM 2048
#define B_DIM 16
#define I_DIM 256
#define H_DIM 256
#define O_DIM 8
#define N_DIM (H_DIM * O_DIM)   // 2048 flattened (h,o); n%8=o, n/8=h
#define M_DIM (T_DIM * B_DIM)   // 32768 flattened (t,b) rows
#define OUT_ELEMS (M_DIM * O_DIM)       // 262144
#define NBLOCKS_N 16                    // N_DIM / 128
#define PARTIALS_BYTES ((size_t)NBLOCKS_N * OUT_ELEMS * 4)  // 16 MB

#define LDS_STRIDE 264          // 256 + 8 pad elements: breaks 512B-stride bank collision

typedef __attribute__((ext_vector_type(16))) __bf16 v16bf;
typedef __attribute__((ext_vector_type(8)))  __bf16 bf16x8;
typedef __attribute__((ext_vector_type(4)))  __bf16 bf16x4;
typedef __attribute__((ext_vector_type(8)))  float  f32x8;
typedef __attribute__((ext_vector_type(4)))  float  f32x4;
typedef __attribute__((ext_vector_type(8)))  float  v8f;

// Packed-native split of fp32 into bf16 hi + bf16 lo (RNE both steps).
__device__ __forceinline__ void split8(f32x8 v, bf16x8& hi, bf16x8& lo) {
  hi = __builtin_convertvector(v, bf16x8);          // v_cvt_pk_bf16_f32 x4
  f32x8 hf = __builtin_convertvector(hi, f32x8);
  lo = __builtin_convertvector(v - hf, bf16x8);
}

__device__ __forceinline__ v16bf cat16(bf16x8 a, bf16x8 b) {
  return __builtin_shufflevector(a, b, 0, 1, 2, 3, 4, 5, 6, 7,
                                       8, 9, 10, 11, 12, 13, 14, 15);
}

// Fused GEMM (bf16x3-split WMMA) + (alpha==1) elementwise epilogue (+ fused
// readout partials when use_partials!=0).
// Block: 256 threads = 8 waves. Each wave owns one 16-column tile, holds weight
// hi/lo fragments in registers across 8 M-tiles. x for tile mi+1 is loaded into
// registers during tile mi's WMMA phase (software pipeline); the bf16 hi/lo
// split lives in LDS so waves read A fragments as plain ds_load_b128.
// grid = (M_DIM/128, NBLOCKS_N) = (256, 16)
__global__ __launch_bounds__(256) void k_gemm_act(
    const float* __restrict__ x,      // (T,B,I)
    const float* __restrict__ Bw,     // (O,H,I)
    const float* __restrict__ Bb,     // (O,H)
    const float* __restrict__ noise,  // (T,B,H,O)
    const float* __restrict__ sigma,  // (H,O) -> flat index == n
    const float* __restrict__ Ww,     // (O,H) readout weights (fused fast path)
    const float* __restrict__ alpha_p,
    float* __restrict__ act,          // (T,B,H,O): activity (a==1) or tot (else)
    float* __restrict__ partials,     // (NBLOCKS_N, T*B, O) readout partials
    int use_partials)
{
  __shared__ __bf16 xhi[16 * LDS_STRIDE];   // 8.25 KB
  __shared__ __bf16 xlo[16 * LDS_STRIDE];   // 8.25 KB
  __shared__ float  red[8 * 128];           // 4 KB cross-wave readout partials

  const int tid  = threadIdx.x;
  const int wave = tid >> 5;
  const int lane = tid & 31;
  const int half = lane >> 4;         // K half select (A/B), row half select (C)
  const int cl   = lane & 15;         // A row within tile == C/D column within tile

  const int n0  = blockIdx.y * 128 + wave * 16;
  const int col = n0 + cl;            // global column n = h*8 + o
  const size_t wrow = ((size_t)(col & 7) * H_DIM + (size_t)(col >> 3)) * I_DIM;

  // ---- Weight fragments for all 8 K-chunks (hi/lo bf16), kept in VGPRs.
  v16bf wh[8], wl[8];
#pragma unroll
  for (int kc = 0; kc < 8; ++kc) {
    const float* wp = Bw + wrow + kc * 32 + half * 8;
    f32x8 g0 = *(const f32x8*)(wp);        // K = kc*32 + half*8 + [0..7]
    f32x8 g1 = *(const f32x8*)(wp + 16);   // K = +16
    bf16x8 h0, l0, h1, l1;
    split8(g0, h0, l0);
    split8(g1, h1, l1);
    wh[kc] = cat16(h0, h1);
    wl[kc] = cat16(l0, l1);
  }

  const float bias = Bb[(size_t)(col & 7) * H_DIM + (size_t)(col >> 3)];
  const float sg   = sigma[col];
  const float wv   = Ww[(size_t)(col & 7) * H_DIM + (size_t)(col >> 3)];
  const float a    = alpha_p[0];
  const bool  fast = (a == 1.0f);     // alpha==1 -> scan degenerates, fuse here
  const bool  fuse_ro = fast && (use_partials != 0);

  const int rowoff = cl * LDS_STRIDE;

  // Per-thread staging slice: 16 floats as 4x float4, coalesced.
  const int s_el  = tid * 4;                 // first element of each float4 group
  const int s_row0 = s_el >> 8;              // row advances by 4 per 1024 elements
  const int s_c   = s_el & 255;

  // Software pipeline: preload m-tile 0 into registers.
  f32x4 xr[4];
  {
    const float* xsrc = x + (size_t)(blockIdx.x * 8) * 16 * I_DIM;
#pragma unroll
    for (int kq = 0; kq < 4; ++kq) xr[kq] = *(const f32x4*)(xsrc + s_el + 1024 * kq);
  }

  for (int mi = 0; mi < 8; ++mi) {
    const int m0 = (blockIdx.x * 8 + mi) * 16;   // 16 rows == all 16 b at one t

    __syncthreads();   // protects xhi/xlo (and red) reuse
    // Convert the in-register x tile to bf16 hi/lo and stage into LDS.
#pragma unroll
    for (int kq = 0; kq < 4; ++kq) {
      f32x4 v = xr[kq];
      bf16x4 h = __builtin_convertvector(v, bf16x4);
      f32x4 hf = __builtin_convertvector(h, f32x4);
      bf16x4 l = __builtin_convertvector(v - hf, bf16x4);
      const int row = s_row0 + kq * 4;
      *(bf16x4*)(xhi + row * LDS_STRIDE + s_c) = h;
      *(bf16x4*)(xlo + row * LDS_STRIDE + s_c) = l;
    }
    __syncthreads();

    // Issue next tile's global loads now; they complete during the WMMA phase.
    if (mi < 7) {
      const float* xnext = x + (size_t)(m0 + 16) * I_DIM;
#pragma unroll
      for (int kq = 0; kq < 4; ++kq) xr[kq] = *(const f32x4*)(xnext + s_el + 1024 * kq);
    }

    v8f acc = {};
#pragma unroll
    for (int kc = 0; kc < 8; ++kc) {
      const int ko = kc * 32 + half * 8;
      bf16x8 a0 = *(const bf16x8*)(xhi + rowoff + ko);        // ds_load_b128
      bf16x8 a1 = *(const bf16x8*)(xhi + rowoff + ko + 16);
      bf16x8 b0 = *(const bf16x8*)(xlo + rowoff + ko);
      bf16x8 b1 = *(const bf16x8*)(xlo + rowoff + ko + 16);
      v16bf ah = cat16(a0, a1);
      v16bf al = cat16(b0, b1);
      // bf16x3 split-product: near-fp32 accuracy at matrix-core rate
      acc = __builtin_amdgcn_wmma_f32_16x16x32_bf16(false, ah, false, wh[kc],
                                                    (short)0, acc, false, false);
      acc = __builtin_amdgcn_wmma_f32_16x16x32_bf16(false, ah, false, wl[kc],
                                                    (short)0, acc, false, false);
      acc = __builtin_amdgcn_wmma_f32_16x16x32_bf16(false, al, false, wh[kc],
                                                    (short)0, acc, false, false);
    }

    if (fast) {
      float av[8];
#pragma unroll
      for (int j = 0; j < 8; ++j) {
        const size_t idx = (size_t)(m0 + half * 8 + j) * N_DIM + col;
        const float s = acc[j] + bias + sg * noise[idx];
        const float t = tanhf(s);
        act[idx] = t;
        av[j] = t * wv;                       // readout partial act * Ww[o,h]
      }
      if (fuse_ro) {
        // Lanes l and l+8 share o (cols n0..n0+15 -> o = l&7); pair-reduce,
        // then cross-wave reduce in LDS, then deterministic partial store.
#pragma unroll
        for (int j = 0; j < 8; ++j) {
          const float r = av[j] + __shfl_xor(av[j], 8, 32);
          if ((lane & 8) == 0)
            red[wave * 128 + (half * 8 + j) * 8 + (lane & 7)] = r;
        }
        __syncthreads();
        if (tid < 128) {                       // tid = row*8 + o
          float s = 0.0f;
#pragma unroll
          for (int w = 0; w < 8; ++w) s += red[w * 128 + tid];
          const int row = tid >> 3;
          partials[(size_t)blockIdx.y * OUT_ELEMS +
                   (size_t)(m0 + row) * O_DIM + (tid & 7)] = s;
        }
      }
    } else {
#pragma unroll
      for (int j = 0; j < 8; ++j) {
        const size_t idx = (size_t)(m0 + half * 8 + j) * N_DIM + col;
        act[idx] = acc[j] + bias;     // tot; scan kernel finishes it
      }
    }
  }
}

// General-alpha sequential scan (no-op when alpha==1). One thread per (b,h,o),
// fully coalesced across the 128 KB t-slab; act holds tot on entry, tanh(state) on exit.
__global__ __launch_bounds__(256) void k_scan(
    const float* __restrict__ noise, const float* __restrict__ sigma,
    const float* __restrict__ alpha_p, float* __restrict__ act)
{
  const float a = alpha_p[0];
  if (a == 1.0f) return;
  const int idx = blockIdx.x * 256 + threadIdx.x;      // (b,h,o) flat, 0..32767
  const float sg = sigma[idx & (N_DIM - 1)];
  const float oma = 1.0f - a;
  float st = 0.0f;
  for (int t = 0; t < T_DIM; ++t) {
    const size_t p = (size_t)t * (B_DIM * N_DIM) + idx;
    st = a * act[p] + oma * st + sg * noise[p];
    act[p] = tanhf(st);
  }
}

// Fast-path finisher: out = Wb + sum over the 16 n-block partials.
// Deterministic (fixed summation order). No-op when alpha != 1.
__global__ __launch_bounds__(256) void k_finish(
    const float* __restrict__ partials, const float* __restrict__ Wb,
    const float* __restrict__ alpha_p, float* __restrict__ out)
{
  if (alpha_p[0] != 1.0f) return;
  const int i = blockIdx.x * 256 + threadIdx.x;        // 0..OUT_ELEMS-1
  float s = Wb[i & 7];
#pragma unroll
  for (int ny = 0; ny < NBLOCKS_N; ++ny)
    s += partials[(size_t)ny * OUT_ELEMS + i];
  out[i] = s;
}

// Unfused readout from activity: out[t,b,o] = sum_h act[t,b,h,o]*Ww[o,h] + Wb[o].
// When `gated` != 0, only runs for alpha != 1 (fast path handled by k_finish).
__global__ __launch_bounds__(256) void k_readout(
    const float* __restrict__ act, const float* __restrict__ Ww,
    const float* __restrict__ Wb, const float* __restrict__ alpha_p,
    int gated, float* __restrict__ out)
{
  if (gated && alpha_p[0] == 1.0f) return;
  __shared__ float red[256];
  const int tid = threadIdx.x;
  const int o   = tid & 7;
  const size_t base = (size_t)blockIdx.x * N_DIM;
  float p = 0.0f;
#pragma unroll
  for (int k = 0; k < 8; ++k) {
    const int e = tid + 256 * k;     // e%8 == tid%8 == o for all k
    const int h = e >> 3;
    p += act[base + e] * Ww[(size_t)o * H_DIM + h];
  }
  red[tid] = p;
  __syncthreads();
  for (int s = 128; s >= 8; s >>= 1) {
    if (tid < s) red[tid] += red[tid + s];
    __syncthreads();
  }
  if (tid < 8) out[(size_t)blockIdx.x * O_DIM + tid] = red[tid] + Wb[tid];
}

extern "C" void kernel_launch(void* const* d_in, const int* in_sizes, int n_in,
                              void* d_out, int out_size, void* d_ws, size_t ws_size,
                              hipStream_t stream) {
  (void)in_sizes; (void)n_in; (void)out_size;
  const float* x     = (const float*)d_in[0];
  const float* noise = (const float*)d_in[1];
  const float* sigma = (const float*)d_in[2];
  const float* Bw    = (const float*)d_in[3];
  const float* Bb    = (const float*)d_in[4];
  const float* Ww    = (const float*)d_in[5];
  const float* Wb    = (const float*)d_in[6];
  const float* alpha = (const float*)d_in[7];

  float* out = (float*)d_out;                        // (T,B,O) = 262144 floats
  float* act = out + (size_t)OUT_ELEMS;              // (T,B,H,O) activity region
  float* partials = (float*)d_ws;

  const int use_partials = (d_ws != nullptr && ws_size >= PARTIALS_BYTES) ? 1 : 0;

  dim3 g1(M_DIM / 128, NBLOCKS_N);                   // (256, 16)
  k_gemm_act<<<g1, 256, 0, stream>>>(x, Bw, Bb, noise, sigma, Ww, alpha,
                                     act, partials, use_partials);
  k_scan<<<(B_DIM * N_DIM) / 256, 256, 0, stream>>>(noise, sigma, alpha, act);
  if (use_partials) {
    k_finish<<<OUT_ELEMS / 256, 256, 0, stream>>>(partials, Wb, alpha, out);
    k_readout<<<M_DIM, 256, 0, stream>>>(act, Ww, Wb, alpha, 1, out);
  } else {
    k_readout<<<M_DIM, 256, 0, stream>>>(act, Ww, Wb, alpha, 0, out);
  }
}
